// DynChunking_13709535609070
// MI455X (gfx1250) — matrix operand order repaired and checked by reading, not compiled, not run on hardware
//
#include <hip/hip_runtime.h>

#define B_  8
#define T_  2048
#define C_  1024
#define BT_ (B_ * T_)

typedef __attribute__((ext_vector_type(16))) __bf16 v16bf;
typedef __attribute__((ext_vector_type(8)))  float  v8f;

__device__ __forceinline__ __bf16 us_to_bf(unsigned short u) {
    return __builtin_bit_cast(__bf16, u);
}
__device__ __forceinline__ unsigned short bf_to_us(__bf16 b) {
    return __builtin_bit_cast(unsigned short, b);
}
// Unpack 4 packed uints (8 bf16) into fragment elements [base, base+8)
__device__ __forceinline__ void unpack8(const unsigned int* __restrict__ p,
                                        v16bf& v, int base) {
    #pragma unroll
    for (int j = 0; j < 4; ++j) {
        unsigned int u = p[j];
        v[base + 2 * j]     = us_to_bf((unsigned short)(u & 0xffffu));
        v[base + 2 * j + 1] = us_to_bf((unsigned short)(u >> 16));
    }
}

// ---------------------------------------------------------------------------
// Kernel 0: one-shot bf16 hi/lo split of W (hoists all W conversions out of
// the GEMM inner loop).
// ---------------------------------------------------------------------------
__global__ __launch_bounds__(256) void split_w_kernel(
    const float* __restrict__ W,
    unsigned short* __restrict__ whi, unsigned short* __restrict__ wlo)
{
    int i = blockIdx.x * 256 + threadIdx.x;
    if (i < 2 * C_ * C_) {
        float f = W[i];
        __bf16 h = (__bf16)f;
        __bf16 l = (__bf16)(f - (float)h);
        whi[i] = bf_to_us(h);
        wlo[i] = bf_to_us(l);
    }
}

// ---------------------------------------------------------------------------
// Kernel 1: WMMA GEMM band kernel, 32 rows/block (2 row-tiles per wave).
// Grid: (T/32, B). Block: 256 threads = 8 wave32 waves.
// Waves 0-3: k = W[0:C] @ x, rows t0-1 .. t0+30 (shifted by -1)
// Waves 4-7: q = W[C:2C] @ x, rows t0 .. t0+31
// bf16 hi/lo split (3 WMMAs per tile) => 6 WMMAs per K-step per wave.
// Reduces to Skk[t], Sqq[t], qk[t] = q_t . k_{t-1}. No atomics.
// ---------------------------------------------------------------------------
__global__ __launch_bounds__(256) void kq_stats_kernel(
    const float* __restrict__ x,
    const unsigned short* __restrict__ whi, const unsigned short* __restrict__ wlo,
    float* __restrict__ Skk, float* __restrict__ Sqq, float* __restrict__ qkO)
{
    __shared__ __align__(16) unsigned short xhi[33 * 1024];
    __shared__ __align__(16) unsigned short xlo[33 * 1024];
    __shared__ float ktile[4][2][256];
    __shared__ float partk[4][32], partq[4][32], partqk[4][32];

    const int tid  = threadIdx.x;
    const int wave = tid >> 5;
    const int lane = tid & 31;
    const int t0   = blockIdx.x * 32;
    const int b    = blockIdx.y;

    // Load x rows t0-1 .. t0+31 (clamp row -1; its results are never used)
    for (int i = tid; i < 33 * 1024; i += 256) {
        int r = i >> 10, c = i & 1023;
        int gr = t0 - 1 + r; if (gr < 0) gr = 0;
        float f = x[((size_t)b * T_ + gr) * C_ + c];
        __bf16 h = (__bf16)f;
        __bf16 l = (__bf16)(f - (float)h);
        xhi[i] = bf_to_us(h);
        xlo[i] = bf_to_us(l);
    }
    __syncthreads();

    const bool isq   = (wave >= 4);
    const int  ws    = wave & 3;
    const int  m     = lane & 15;            // N column / M row within tile
    const int  hi8   = (lane & 16) ? 8 : 0;  // C/D: lanes 16-31 hold M+8
    const int  koffA = (lane & 16) ? 8 : 0;  // A: lane halves K offset
    const int  koffB = (lane & 16) ? 8 : 0;  // B: lane halves K offset
    const int  rowoff = isq ? 1 : 0;
    const unsigned int* pxh0 = (const unsigned int*)&xhi[(m + rowoff) * 1024];
    const unsigned int* pxl0 = (const unsigned int*)&xlo[(m + rowoff) * 1024];
    const unsigned int* pxh1 = pxh0 + 16 * 512;   // +16 rows (512 uints/row)
    const unsigned int* pxl1 = pxl0 + 16 * 512;

    float ssp[2][8] = {}; // per-lane |.|^2 partials (per row-tile)
    float qkp[2][8] = {}; // per-lane q.k_prev partials

    for (int c0 = 0; c0 < C_; c0 += 64) {
        const int chan = (isq ? C_ : 0) + c0 + ws * 16 + m;
        const unsigned int* wh32 = (const unsigned int*)(whi + (size_t)chan * C_);
        const unsigned int* wl32 = (const unsigned int*)(wlo + (size_t)chan * C_);

        v8f acc0 = {}; v8f acc1 = {};
        for (int kb = 0; kb < C_; kb += 32) {
            const int ia = (kb + koffA) >> 1;
            const int ib = (kb + koffB) >> 1;
            v16bf a0h, a0l, a1h, a1l, bh, bl;
            unpack8(pxh0 + ia, a0h, 0); unpack8(pxh0 + ia + 8, a0h, 8);
            unpack8(pxl0 + ia, a0l, 0); unpack8(pxl0 + ia + 8, a0l, 8);
            unpack8(pxh1 + ia, a1h, 0); unpack8(pxh1 + ia + 8, a1h, 8);
            unpack8(pxl1 + ia, a1l, 0); unpack8(pxl1 + ia + 8, a1l, 8);
            unpack8(wh32 + ib, bh, 0);  unpack8(wh32 + ib + 8, bh, 8);
            unpack8(wl32 + ib, bl, 0);  unpack8(wl32 + ib + 8, bl, 8);

            acc0 = __builtin_amdgcn_wmma_f32_16x16x32_bf16(false, a0h, false, bh,
                                                           (short)0, acc0, false, false);
            acc0 = __builtin_amdgcn_wmma_f32_16x16x32_bf16(false, a0h, false, bl,
                                                           (short)0, acc0, false, false);
            acc0 = __builtin_amdgcn_wmma_f32_16x16x32_bf16(false, a0l, false, bh,
                                                           (short)0, acc0, false, false);
            acc1 = __builtin_amdgcn_wmma_f32_16x16x32_bf16(false, a1h, false, bh,
                                                           (short)0, acc1, false, false);
            acc1 = __builtin_amdgcn_wmma_f32_16x16x32_bf16(false, a1h, false, bl,
                                                           (short)0, acc1, false, false);
            acc1 = __builtin_amdgcn_wmma_f32_16x16x32_bf16(false, a1l, false, bh,
                                                           (short)0, acc1, false, false);
        }

        if (!isq) {
            #pragma unroll
            for (int r = 0; r < 8; ++r) {
                ktile[ws][0][(r + hi8) * 16 + m] = acc0[r];
                ktile[ws][1][(r + hi8) * 16 + m] = acc1[r];
            }
        }
        #pragma unroll
        for (int r = 0; r < 8; ++r) {
            ssp[0][r] += acc0[r] * acc0[r];
            ssp[1][r] += acc1[r] * acc1[r];
        }
        __syncthreads();                       // k tiles visible
        if (isq) {
            #pragma unroll
            for (int r = 0; r < 8; ++r) {
                qkp[0][r] += acc0[r] * ktile[ws][0][(r + hi8) * 16 + m];
                qkp[1][r] += acc1[r] * ktile[ws][1][(r + hi8) * 16 + m];
            }
        }
        __syncthreads();                       // safe to overwrite ktile
    }

    // Reduce across the 16 lanes of each half-wave; stash per wave.
    #pragma unroll
    for (int j = 0; j < 2; ++j) {
        #pragma unroll
        for (int r = 0; r < 8; ++r) {
            float v = ssp[j][r];
            for (int d = 1; d < 16; d <<= 1) v += __shfl_xor(v, d, 32);
            float u = qkp[j][r];
            for (int d = 1; d < 16; d <<= 1) u += __shfl_xor(u, d, 32);
            if (m == 0) {
                int M = 16 * j + r + hi8;
                if (!isq) partk[ws][M] = v;
                else      { partq[ws][M] = v; partqk[ws][M] = u; }
            }
        }
    }
    __syncthreads();

    // Fixed-order cross-wave combine (deterministic), one row per thread.
    if (tid < 32) {
        int M = tid;
        float skk = partk[0][M] + partk[1][M] + partk[2][M] + partk[3][M];
        float sqq = partq[0][M] + partq[1][M] + partq[2][M] + partq[3][M];
        float qks = partqk[0][M] + partqk[1][M] + partqk[2][M] + partqk[3][M];
        int tr = t0 - 1 + M;
        if (tr >= 0) Skk[(size_t)b * T_ + tr] = skk;   // row T-1 never needed
        Sqq[(size_t)b * T_ + t0 + M] = sqq;
        qkO[(size_t)b * T_ + t0 + M] = qks;
    }
}

// ---------------------------------------------------------------------------
// Kernel 2: p / bt / stable partition per batch row. Grid: (B). Block: 256.
// ---------------------------------------------------------------------------
__global__ __launch_bounds__(256) void finalize_kernel(
    const float* __restrict__ Skk, const float* __restrict__ Sqq,
    const float* __restrict__ qk,
    float* __restrict__ out_p, float* __restrict__ out_bt,
    float* __restrict__ out_Pc, float* __restrict__ out_mask,
    float* __restrict__ out_gidx, int* __restrict__ sel, int* __restrict__ counts)
{
    __shared__ float lds_p[T_];
    __shared__ int   lds_sel[T_];
    __shared__ int   scan[256];

    const int b = blockIdx.x, tid = threadIdx.x;
    int keepf[8], lpref[8];
    int cnt = 0;
    #pragma unroll
    for (int j = 0; j < 8; ++j) {
        int t = tid * 8 + j;
        float p;
        if (t == 0) {
            p = 1.0f;
        } else {
            float nq = sqrtf(Sqq[(size_t)b * T_ + t]) + 1e-8f;
            float nk = sqrtf(Skk[(size_t)b * T_ + t - 1]) + 1e-8f;
            float cs = qk[(size_t)b * T_ + t] / (nq * nk);
            p = 0.5f * (1.0f - cs);
        }
        int bt = (t == 0) ? 1 : (p >= 0.5f ? 1 : 0);
        lds_p[t] = p;
        out_p [(size_t)b * T_ + t] = p;
        out_bt[(size_t)b * T_ + t] = (float)bt;
        keepf[j] = bt;
        lpref[j] = cnt;
        cnt += bt;
    }
    scan[tid] = cnt;
    __syncthreads();
    for (int off = 1; off < 256; off <<= 1) {
        int v = scan[tid];
        int u = (tid >= off) ? scan[tid - off] : 0;
        __syncthreads();
        scan[tid] = v + u;
        __syncthreads();
    }
    const int total = scan[255];
    const int excl  = scan[tid] - cnt;

    #pragma unroll
    for (int j = 0; j < 8; ++j) {
        int t  = tid * 8 + j;
        int kb = excl + lpref[j];                         // kept before t
        int pos = keepf[j] ? kb : (total + (t - kb));     // stable partition
        lds_sel[pos] = t;
    }
    if (tid == 0) counts[b] = total;
    __syncthreads();

    #pragma unroll
    for (int j = 0; j < 8; ++j) {
        int i = tid * 8 + j;
        int s = lds_sel[i];
        int mk = (i < total) ? 1 : 0;
        sel[(size_t)b * T_ + i] = s;
        out_Pc  [(size_t)b * T_ + i] = mk ? lds_p[s] : 0.0f;
        out_mask[(size_t)b * T_ + i] = (float)mk;
        out_gidx[(size_t)b * T_ + i] = mk ? (float)s : 0.0f;
    }
}

// ---------------------------------------------------------------------------
// Kernel 3: bandwidth-bound row gather of x. Grid: (T, B). Block: 256 (float4).
// ---------------------------------------------------------------------------
__global__ __launch_bounds__(256) void gather_kernel(
    const float* __restrict__ x, const int* __restrict__ sel,
    const int* __restrict__ counts, float* __restrict__ out_x)
{
    const int i = blockIdx.x, b = blockIdx.y, tid = threadIdx.x;
    const int total = counts[b];
    float4 v = make_float4(0.f, 0.f, 0.f, 0.f);
    if (i < total) {
        int s = sel[(size_t)b * T_ + i];
        v = ((const float4*)(x + ((size_t)b * T_ + s) * C_))[tid];
    }
    ((float4*)(out_x + ((size_t)b * T_ + i) * C_))[tid] = v;
}

// ---------------------------------------------------------------------------
extern "C" void kernel_launch(void* const* d_in, const int* in_sizes, int n_in,
                              void* d_out, int out_size, void* d_ws, size_t ws_size,
                              hipStream_t stream) {
    (void)in_sizes; (void)n_in; (void)out_size; (void)ws_size;
    const float* x = (const float*)d_in[0];   // (8, 2048, 1024) fp32
    const float* W = (const float*)d_in[1];   // (2048, 1024)    fp32

    float* out    = (float*)d_out;
    float* ox     = out;                                  // x_chunks  (B,T,C)
    float* oP     = out + (size_t)BT_ * C_;               // P_chunks  (B,T)
    float* omask  = oP + BT_;                             // mask_ds   (B,T)
    float* op     = omask + BT_;                          // p         (B,T)
    float* obt    = op + BT_;                             // bt        (B,T)
    float* ogidx  = obt + BT_;                            // gather_idx(B,T)

    float* ws     = (float*)d_ws;
    float* Skk    = ws;
    float* Sqq    = ws + BT_;
    float* qk     = ws + 2 * BT_;
    int*   sel    = (int*)(ws + 3 * BT_);
    int*   counts = sel + BT_;

    // Use the (not-yet-written) x_chunks output region as scratch for the
    // bf16 hi/lo split of W (8 MB of 64 MB); gather_kernel overwrites it last.
    unsigned short* whi = (unsigned short*)ox;            // 4 MB
    unsigned short* wlo = whi + (size_t)2 * C_ * C_;      // 4 MB

    split_w_kernel<<<(2 * C_ * C_ + 255) / 256, 256, 0, stream>>>(W, whi, wlo);

    dim3 g1(T_ / 32, B_);
    kq_stats_kernel<<<g1, 256, 0, stream>>>(x, whi, wlo, Skk, Sqq, qk);

    finalize_kernel<<<B_, 256, 0, stream>>>(Skk, Sqq, qk, op, obt, oP, omask,
                                            ogidx, sel, counts);

    dim3 g3(T_, B_);
    gather_kernel<<<g3, 256, 0, stream>>>(x, sel, counts, ox);
}
